// MemoryReader_57251914055875
// MI455X (gfx1250) — matrix operand order repaired and checked by reading, not compiled, not run on hardware
//
#include <hip/hip_runtime.h>
#include <hip/hip_bf16.h>

// ---------------------------------------------------------------------------
// MemoryReader for MI455X (gfx1250, wave32, WMMA bf16 path)
// ---------------------------------------------------------------------------
#define BB 8
#define MM 8192
#define SS 128
#define DD 512
#define QQ 32
#define HH 8
#define HD 64
// scale = 1/sqrt(HD) = 0.125

#define ASYNC_COPY 1   // CDNA5 global_load_async_to_lds_b128 for A-tile staging

#if defined(__has_builtin)
#if __has_builtin(__builtin_amdgcn_tensor_load_to_lds)
#define HAVE_TDM 1
#endif
#endif
#ifndef HAVE_TDM
#define HAVE_TDM 0
#endif

typedef __bf16 bf16;
typedef __attribute__((ext_vector_type(16))) __bf16 v16bf;
typedef __attribute__((ext_vector_type(8)))  float  v8f;
typedef unsigned int __attribute__((ext_vector_type(4))) v4u;
typedef int __attribute__((ext_vector_type(4))) v4i;
typedef int __attribute__((ext_vector_type(8))) v8i;

union Frag {            // one WMMA 16-bit operand: 16 bf16 = 8 dwords per lane
    v16bf   v;
    unsigned u[8];
};

__device__ __forceinline__ unsigned pack_bf16(float a, float b) {
    union { __bf16 h[2]; unsigned u; } x;
    x.h[0] = (__bf16)a; x.h[1] = (__bf16)b;
    return x.u;
}

__device__ __forceinline__ v8f wmma_bf16(const Frag& a, const Frag& b, v8f c) {
    // D = A(16x32 bf16) x B(32x16 bf16) + C(16x16 f32)
    return __builtin_amdgcn_wmma_f32_16x16x32_bf16(false, a.v, false, b.v,
                                                   (short)0, c, false, false);
}

__device__ __forceinline__ v8f zero8() {
    v8f z; for (int i = 0; i < 8; ++i) z[i] = 0.0f; return z;
}

// A-fragment k index per ISA 16-bit A 16x32 layout (dword j, half = lane/16)
__device__ __forceinline__ int a_kidx(int j, int half) {
    return (j < 4) ? (half * 8 + 2 * j) : (16 + half * 8 + 2 * (j - 4));
}
// B-fragment k index: same k order as A fragment
__device__ __forceinline__ int b_kidx(int j, int half) {
    return (j < 4) ? (half * 8 + 2 * j) : (16 + half * 8 + 2 * (j - 4));
}

// ---------------------------------------------------------------------------
// Kernel 1: pooled context, queries (output), gates (output), scaled q-heads
// grid = B, block = 256
// ---------------------------------------------------------------------------
__global__ void prep_kernel(const float* __restrict__ context,
                            const float* __restrict__ qp,
                            const float* __restrict__ ctx_w, const float* __restrict__ ctx_b,
                            const float* __restrict__ in_w,  const float* __restrict__ in_b,
                            const float* __restrict__ gate_w, const float* __restrict__ gate_b,
                            float* __restrict__ out_gates,
                            float* __restrict__ out_queries,
                            bf16* __restrict__ qs) {
    __shared__ float pooled[DD];
    __shared__ float cvec[DD];
    const int b = blockIdx.x, t = threadIdx.x;

    for (int c = t; c < DD; c += 256) {
        float s = 0.f;
        const float* p = context + ((size_t)b * SS) * DD + c;
        for (int ss = 0; ss < SS; ++ss) s += p[(size_t)ss * DD];
        pooled[c] = s * (1.0f / SS);
    }
    __syncthreads();
    for (int c = t; c < DD; c += 256) {
        float s = ctx_b[c];
        const float* w = ctx_w + (size_t)c * DD;
        for (int d = 0; d < DD; ++d) s += pooled[d] * w[d];
        cvec[c] = s;
    }
    __syncthreads();
    for (int e = t; e < QQ * DD; e += 256) {
        int q = e / DD, d = e % DD;
        out_queries[((size_t)b * QQ + q) * DD + d] = qp[e] + cvec[d];
    }
    if (t < QQ) {
        float s = gate_b[t];
        const float* w = gate_w + (size_t)t * DD;
        for (int d = 0; d < DD; ++d) s += pooled[d] * w[d];
        out_gates[b * QQ + t] = 1.0f / (1.0f + __expf(-s));
    }
    // q heads, pre-scaled by 1/sqrt(HD): qs[(b*H+h)][q][hd]
    for (int e = t; e < QQ * DD; e += 256) {
        int q = e / DD, c = e % DD;
        const float* w = in_w + (size_t)c * DD;      // wq row c
        float s = in_b[c];
        for (int d = 0; d < DD; ++d) s += (qp[q * DD + d] + cvec[d]) * w[d];
        s *= 0.125f;
        int h = c >> 6, hd = c & 63;
        qs[(((size_t)b * HH + h) * QQ + q) * HD + hd] = (bf16)s;
    }
}

// ---------------------------------------------------------------------------
// Kernel 2: K/V projection  C[65536,1024] = memory[65536,512] @ Wkv^T
// cols 0..511 -> K[bh][m][hd], 512..1023 -> Vt[bh][hd][m]
// grid = (B*M/128, 8), block = 256 (8 waves, 4x2 wave grid, 32x64 per wave)
// Dynamic LDS layout (66048 bytes):
//   [0,16K)      stageA f32 128x32     (A tile, async-loaded)
//   [16K,32K)    stageB f32 128x32     (B tile, TDM- or async-loaded)
//   [32K,41K)    tA bf16 128x36 padded
//   [41K,50K)    tB bf16 128x36 padded
//   epilogue:    outT f32 128x129 overlaps everything (66048B)
// ---------------------------------------------------------------------------
#define LDA 36
#define KV_SMEM (128 * 129 * 4)

__global__ void __launch_bounds__(256)
kv_proj_kernel(const float* __restrict__ memory,
               const float* __restrict__ in_w,
               const float* __restrict__ in_b,
               bf16* __restrict__ Kbuf,
               bf16* __restrict__ Vt) {
    extern __shared__ char smem[];
    float*    stageA = (float*)smem;
    float*    stageB = (float*)(smem + 16384);
    unsigned* tA32   = (unsigned*)(smem + 32768);
    unsigned* tB32   = (unsigned*)(smem + 32768 + 9216);
    float*    outT   = (float*)smem;                 // epilogue reuse

    const int t = threadIdx.x;
    const int lane = t & 31, wave = t >> 5;
    const int wm = wave >> 1, wn = wave & 1;
    const int half = lane >> 4, l16 = lane & 15;
    const size_t row0 = (size_t)blockIdx.x * 128;    // global row = b*M + m
    const int col0 = blockIdx.y * 128;               // output channel tile

    v8f acc[2][4];
    for (int i = 0; i < 2; ++i)
        for (int j = 0; j < 4; ++j) acc[i][j] = zero8();

    for (int k0 = 0; k0 < DD; k0 += 32) {
#if HAVE_TDM
        if (wave == 0) {
            // Tensor Data Mover: 2D tile (32 x 128 rows, stride 512 f32) -> LDS
            unsigned long long gaddr = (unsigned long long)(const void*)
                (in_w + (size_t)(DD + col0) * DD + k0);
            unsigned ldsb = (unsigned)(size_t)stageB;
            v4u g0 = { 1u,                                  // count = 1 (valid)
                       ldsb,                                // lds_addr
                       (unsigned)(gaddr & 0xffffffffu),     // global_addr lo
                       (unsigned)((gaddr >> 32) & 0x01ffffffu) | (2u << 30) }; // hi | type=2
            v8i g1 = { (int)(2u << 16),                     // data_size = 4B
                       (int)(32u << 16),                    // tensor_dim0 = 32
                       (int)(128u << 16),                   // tensor_dim1 = 128
                       (int)(32u << 16),                    // tile_dim0 = 32
                       (int)128,                            // tile_dim1 = 128
                       (int)DD,                             // tensor_dim0_stride = 512
                       0, 0 };
            v4i g2 = { 0, 0, 0, 0 };
            v4i g3 = { 0, 0, 0, 0 };
#if __clang_major__ >= 23
            v8i g4 = { 0, 0, 0, 0, 0, 0, 0, 0 };
            __builtin_amdgcn_tensor_load_to_lds(g0, g1, g2, g3, g4, 0);
#else
            __builtin_amdgcn_tensor_load_to_lds(g0, g1, g2, g3, 0);
#endif
            __builtin_amdgcn_s_wait_tensorcnt(0);
        }
#endif
#if ASYNC_COPY
        {   // per-lane async copy of the A (memory) tile, tracked by ASYNCcnt
            unsigned ldsA = (unsigned)(size_t)stageA;
#if !HAVE_TDM
            unsigned ldsB = (unsigned)(size_t)stageB;
#endif
            for (int i = 0; i < 4; ++i) {
                int idx = i * 256 + t;                      // float4 index
                int r = idx >> 3, c4 = idx & 7;
                unsigned long long ga = (unsigned long long)(const void*)
                    (memory + (row0 + r) * (size_t)DD + k0 + c4 * 4);
                unsigned la = ldsA + (unsigned)idx * 16u;
                asm volatile("global_load_async_to_lds_b128 %0, %1, off"
                             :: "v"(la), "v"(ga) : "memory");
#if !HAVE_TDM
                unsigned long long gb = (unsigned long long)(const void*)
                    (in_w + (size_t)(DD + col0 + r) * DD + k0 + c4 * 4);
                unsigned lb = ldsB + (unsigned)idx * 16u;
                asm volatile("global_load_async_to_lds_b128 %0, %1, off"
                             :: "v"(lb), "v"(gb) : "memory");
#endif
            }
            asm volatile("s_wait_asynccnt 0" ::: "memory");
        }
        __syncthreads();
        {   // convert f32 stage tiles -> padded bf16 tiles
            int r = t >> 1, c0 = (t & 1) * 16;
            const float* sa = &stageA[r * 32 + c0];
            const float* sb = &stageB[r * 32 + c0];
            unsigned* da = &tA32[r * (LDA / 2) + (c0 >> 1)];
            unsigned* db = &tB32[r * (LDA / 2) + (c0 >> 1)];
            for (int j = 0; j < 8; ++j) {
                da[j] = pack_bf16(sa[2 * j], sa[2 * j + 1]);
                db[j] = pack_bf16(sb[2 * j], sb[2 * j + 1]);
            }
        }
#else
        {   // fallback: direct global loads + convert
            int r = t >> 1, c0 = (t & 1) * 16;
            const float* sa = memory + (row0 + r) * (size_t)DD + k0 + c0;
            const float* sb = in_w + (size_t)(DD + col0 + r) * DD + k0 + c0;
            unsigned* da = &tA32[r * (LDA / 2) + (c0 >> 1)];
            unsigned* db = &tB32[r * (LDA / 2) + (c0 >> 1)];
            for (int j = 0; j < 8; ++j) {
                da[j] = pack_bf16(sa[2 * j], sa[2 * j + 1]);
                db[j] = pack_bf16(sb[2 * j], sb[2 * j + 1]);
            }
        }
#endif
        __syncthreads();

        Frag af[2], bf[4];
        for (int i = 0; i < 2; ++i) {
            int m = wm * 32 + i * 16 + l16;
            for (int j = 0; j < 8; ++j)
                af[i].u[j] = tA32[m * (LDA / 2) + (a_kidx(j, half) >> 1)];
        }
        for (int i = 0; i < 4; ++i) {
            int n = wn * 64 + i * 16 + l16;
            for (int j = 0; j < 8; ++j)
                bf[i].u[j] = tB32[n * (LDA / 2) + (b_kidx(j, half) >> 1)];
        }
        for (int i = 0; i < 2; ++i)
            for (int j = 0; j < 4; ++j)
                acc[i][j] = wmma_bf16(af[i], bf[j], acc[i][j]);
        __syncthreads();
    }

    // ---- epilogue: accumulators -> padded LDS tile (+bias), coalesced store
    for (int i = 0; i < 2; ++i)
        for (int j = 0; j < 4; ++j) {
            int col = wn * 64 + j * 16 + l16;
            float bias = in_b[DD + col0 + col];
            for (int r = 0; r < 8; ++r) {
                int mrow = wm * 32 + i * 16 + half * 8 + r;
                outT[mrow * 129 + col] = acc[i][j][r] + bias;
            }
        }
    __syncthreads();

    if (col0 < DD) {
        // K part: row-major K[bh][m][hd]; thread = (m = t/2, 64-col segment)
        int m = t >> 1, cs = (t & 1) * 64;
        size_t grow = row0 + m;
        size_t b = grow >> 13, mloc = grow & (MM - 1);
        int h = (col0 + cs) >> 6;
        unsigned* dst = (unsigned*)(Kbuf + (((b * HH + h) * (size_t)MM + mloc) * HD));
        for (int j = 0; j < 32; ++j)
            dst[j] = pack_bf16(outT[m * 129 + cs + 2 * j],
                               outT[m * 129 + cs + 2 * j + 1]);
    } else {
        // V part: transposed Vt[bh][hd][m]; thread = (col = t/2, 64-m segment)
        int c = t >> 1, ms = (t & 1) * 64;
        int cv = col0 - DD + c;
        int h = cv >> 6, hd = cv & 63;
        size_t b = row0 >> 13;
        size_t m0 = (row0 & (MM - 1)) + ms;
        unsigned* dst = (unsigned*)(Vt + (((b * HH + h) * (size_t)HD + hd) * (size_t)MM + m0));
        for (int j = 0; j < 32; ++j)
            dst[j] = pack_bf16(outT[(ms + 2 * j) * 129 + c],
                               outT[(ms + 2 * j + 1) * 129 + c]);
    }
}

// ---------------------------------------------------------------------------
// Kernel 3: scores^T[bh][m][q] = K[m][:] . qs[q][:]  (+mask)
// grid = (M/512, B*H), block = 256 (8 waves * 64 m-rows), K-loop over HD
// ---------------------------------------------------------------------------
__global__ void __launch_bounds__(256)
scores_kernel(const bf16* __restrict__ Kbuf,
              const bf16* __restrict__ qs,
              const unsigned char* __restrict__ mask,
              float* __restrict__ SC) {
    const int bh = blockIdx.y, b = bh >> 3;
    const int t = threadIdx.x, lane = t & 31, wave = t >> 5;
    const int half = lane >> 4, l16 = lane & 15;
    const int m0 = blockIdx.x * 512 + wave * 64;

    const unsigned* Kb = (const unsigned*)(Kbuf + (size_t)bh * MM * HD);
    const unsigned* Qb = (const unsigned*)(qs + (size_t)bh * QQ * HD);

    v8f acc[4][2];
    for (int i = 0; i < 4; ++i)
        for (int j = 0; j < 2; ++j) acc[i][j] = zero8();

    for (int k0 = 0; k0 < HD; k0 += 32) {
        Frag af[4], qf[2];
        for (int i = 0; i < 4; ++i) {
            const unsigned* rowp = Kb + (size_t)(m0 + i * 16 + l16) * (HD / 2) + (k0 >> 1);
            for (int j = 0; j < 8; ++j) af[i].u[j] = rowp[a_kidx(j, half) >> 1];
        }
        for (int i = 0; i < 2; ++i) {
            const unsigned* rowp = Qb + (size_t)(i * 16 + l16) * (HD / 2) + (k0 >> 1);
            for (int j = 0; j < 8; ++j) qf[i].u[j] = rowp[b_kidx(j, half) >> 1];
        }
        for (int i = 0; i < 4; ++i)
            for (int j = 0; j < 2; ++j)
                acc[i][j] = wmma_bf16(af[i], qf[j], acc[i][j]);
    }

    float* SCb = SC + (size_t)bh * MM * QQ;
    for (int i = 0; i < 4; ++i)
        for (int r = 0; r < 8; ++r) {
            int m = m0 + i * 16 + half * 8 + r;
            bool keep = mask[(size_t)b * MM + m] != 0;
            for (int j = 0; j < 2; ++j) {
                int q = j * 16 + l16;
                SCb[(size_t)m * QQ + q] = keep ? acc[i][j][r] : -3.0e38f;
            }
        }
}

// ---------------------------------------------------------------------------
// Kernel 4: softmax over m (per q), write probs bf16 transposed: P[bh][q][m]
// grid = B*H, block = 256 (thread = (q = t&31, m-group = t>>5))
// ---------------------------------------------------------------------------
__global__ void softmax_kernel(const float* __restrict__ SC, bf16* __restrict__ P) {
    const int bh = blockIdx.x, t = threadIdx.x;
    const int q = t & 31, grp = t >> 5;
    __shared__ float red[256];
    __shared__ float mx[32], sm[32];
    __shared__ float tile[128][33];
    const float* S = SC + (size_t)bh * MM * QQ;

    float m_ = -3.4e38f;
    for (int m = grp; m < MM; m += 8) m_ = fmaxf(m_, S[(size_t)m * QQ + q]);
    red[t] = m_;
    __syncthreads();
    if (t < 32) {
        float v = red[t];
        for (int i = 1; i < 8; ++i) v = fmaxf(v, red[t + 32 * i]);
        mx[t] = v;
    }
    __syncthreads();
    const float qmax = mx[q];

    float s_ = 0.f;
    for (int m = grp; m < MM; m += 8) s_ += __expf(S[(size_t)m * QQ + q] - qmax);
    red[t] = s_;
    __syncthreads();
    if (t < 32) {
        float v = red[t];
        for (int i = 1; i < 8; ++i) v += red[t + 32 * i];
        sm[t] = 1.0f / v;
    }
    __syncthreads();
    const float rs = sm[q];

    unsigned* Pb = (unsigned*)P + (size_t)bh * QQ * (MM / 2);
    for (int m0 = 0; m0 < MM; m0 += 128) {
        for (int i = 0; i < 16; ++i) {
            int mloc = i * 8 + grp;
            tile[mloc][q] = __expf(S[(size_t)(m0 + mloc) * QQ + q] - qmax) * rs;
        }
        __syncthreads();
        {
            int qq = t >> 3, seg = t & 7;
            for (int j = 0; j < 8; ++j) {
                int mpair = seg * 8 + j;
                unsigned u = pack_bf16(tile[mpair * 2][qq], tile[mpair * 2 + 1][qq]);
                Pb[(size_t)qq * (MM / 2) + (m0 >> 1) + mpair] = u;
            }
        }
        __syncthreads();
    }
}

// ---------------------------------------------------------------------------
// Kernel 4b: attention output = mean over heads of P
// ---------------------------------------------------------------------------
__global__ void attnmean_kernel(const bf16* __restrict__ P, float* __restrict__ out_att) {
    const size_t i = (size_t)blockIdx.x * 256 + threadIdx.x;   // B*Q*M/2 total
    const unsigned* Pu = (const unsigned*)P;
    const size_t mp = i % (MM / 2);
    const size_t bq = i / (MM / 2);
    const size_t b = bq / QQ, q = bq % QQ;
    float s0 = 0.f, s1 = 0.f;
    for (int h = 0; h < HH; ++h) {
        union { unsigned u; __bf16 h2[2]; } x;
        x.u = Pu[(((b * HH + h) * QQ + q) * (size_t)(MM / 2)) + mp];
        s0 += (float)x.h2[0];
        s1 += (float)x.h2[1];
    }
    out_att[bq * (size_t)MM + 2 * mp]     = s0 * 0.125f;
    out_att[bq * (size_t)MM + 2 * mp + 1] = s1 * 0.125f;
}

// ---------------------------------------------------------------------------
// Kernel 5: ctx[b][q][h*64+hd] = sum_m P[bh][q][m] * V[m][hd]   (split-K)
// grid = B*H, block = 256 (8 waves, each 1024 m), LDS cross-wave reduce
// ---------------------------------------------------------------------------
__global__ void __launch_bounds__(256)
ctx_kernel(const bf16* __restrict__ P, const bf16* __restrict__ Vt,
           bf16* __restrict__ ctx) {
    const int bh = blockIdx.x, b = bh >> 3, h = bh & 7;
    const int t = threadIdx.x, lane = t & 31, wave = t >> 5;
    const int half = lane >> 4, l16 = lane & 15;
    __shared__ float part[8][32][64];    // 64 KB

    const unsigned* Pu = (const unsigned*)P + (size_t)bh * QQ * (MM / 2);
    const unsigned* Vu = (const unsigned*)Vt + (size_t)bh * HD * (MM / 2);

    v8f acc[2][4];
    for (int i = 0; i < 2; ++i)
        for (int j = 0; j < 4; ++j) acc[i][j] = zero8();

    const int mbase = wave * 1024;
    for (int k0 = 0; k0 < 1024; k0 += 32) {
        const int mk = mbase + k0;
        Frag af[2], vf[4];
        for (int i = 0; i < 2; ++i) {
            const unsigned* rowp = Pu + (size_t)(i * 16 + l16) * (MM / 2) + (mk >> 1);
            for (int j = 0; j < 8; ++j) af[i].u[j] = rowp[a_kidx(j, half) >> 1];
        }
        for (int i = 0; i < 4; ++i) {
            const unsigned* rowp = Vu + (size_t)(i * 16 + l16) * (MM / 2) + (mk >> 1);
            for (int j = 0; j < 8; ++j) vf[i].u[j] = rowp[b_kidx(j, half) >> 1];
        }
        for (int i = 0; i < 2; ++i)
            for (int j = 0; j < 4; ++j)
                acc[i][j] = wmma_bf16(af[i], vf[j], acc[i][j]);
    }

    for (int i = 0; i < 2; ++i)
        for (int j = 0; j < 4; ++j)
            for (int r = 0; r < 8; ++r)
                part[wave][i * 16 + half * 8 + r][j * 16 + l16] = acc[i][j][r];
    __syncthreads();

    for (int e = t; e < QQ * HD; e += 256) {
        int q = e >> 6, hd = e & 63;
        float s = 0.f;
        for (int w = 0; w < 8; ++w) s += part[w][q][hd];
        ctx[((size_t)(b * QQ + q)) * DD + h * HD + hd] = (bf16)s;
    }
}

// ---------------------------------------------------------------------------
// Kernel 6: readouts = ctx @ out_w^T + out_b, * gates, LayerNorm -> output
// grid = B*Q/16, block = 256 (8 waves * 64 cols), then wave-shuffle LN
// ---------------------------------------------------------------------------
__global__ void __launch_bounds__(256)
readout_kernel(const bf16* __restrict__ ctx,
               const float* __restrict__ out_w,
               const float* __restrict__ out_b,
               const float* __restrict__ gates,
               const float* __restrict__ ln_g,
               const float* __restrict__ ln_b,
               float* __restrict__ out_read) {
    const int t = threadIdx.x, lane = t & 31, wave = t >> 5;
    const int half = lane >> 4, l16 = lane & 15;
    const int row0 = blockIdx.x * 16;
    __shared__ float R[16][520];

    const unsigned* Cu = (const unsigned*)ctx + (size_t)row0 * (DD / 2);

    v8f acc[4];
    for (int i = 0; i < 4; ++i) acc[i] = zero8();

    for (int k0 = 0; k0 < DD; k0 += 32) {
        Frag af, wf[4];
        {
            const unsigned* rowp = Cu + (size_t)l16 * (DD / 2) + (k0 >> 1);
            for (int j = 0; j < 8; ++j) af.u[j] = rowp[a_kidx(j, half) >> 1];
        }
        for (int i = 0; i < 4; ++i) {
            const float* rowp = out_w + (size_t)(wave * 64 + i * 16 + l16) * DD + k0;
            for (int j = 0; j < 8; ++j) {
                int k = b_kidx(j, half);
                wf[i].u[j] = pack_bf16(rowp[k], rowp[k + 1]);
            }
        }
        for (int i = 0; i < 4; ++i) acc[i] = wmma_bf16(af, wf[i], acc[i]);
    }

    for (int i = 0; i < 4; ++i)
        for (int r = 0; r < 8; ++r) {
            int m = half * 8 + r;
            int col = wave * 64 + i * 16 + l16;
            R[m][col] = (acc[i][r] + out_b[col]) * gates[row0 + m];
        }
    __syncthreads();

    for (int rr = 0; rr < 2; ++rr) {
        int m = wave * 2 + rr;
        float s1 = 0.f, s2 = 0.f;
        for (int c = lane; c < DD; c += 32) {
            float v = R[m][c];
            s1 += v; s2 += v * v;
        }
        for (int off = 16; off; off >>= 1) {
            s1 += __shfl_down(s1, off, 32);
            s2 += __shfl_down(s2, off, 32);
        }
        s1 = __shfl(s1, 0, 32);
        s2 = __shfl(s2, 0, 32);
        float mu = s1 * (1.0f / DD);
        float var = s2 * (1.0f / DD) - mu * mu;
        float rstd = rsqrtf(var + 1e-5f);
        for (int c = lane; c < DD; c += 32) {
            out_read[(size_t)(row0 + m) * DD + c] =
                (R[m][c] - mu) * rstd * ln_g[c] + ln_b[c];
        }
    }
}

// ---------------------------------------------------------------------------
// Host launcher.  Workspace layout (assumes ws_size >= ~193 MB):
//   [0,64MB)    K bf16 [bh][m][hd]   (reused afterwards as P bf16 [bh][q][m])
//   [64,128MB)  Vt bf16 [bh][hd][m]
//   [128,192MB) scores f32 [bh][m][q]
//   [192MB,+)   q-heads bf16, ctx bf16
// ---------------------------------------------------------------------------
extern "C" void kernel_launch(void* const* d_in, const int* in_sizes, int n_in,
                              void* d_out, int out_size, void* d_ws, size_t ws_size,
                              hipStream_t stream) {
    (void)in_sizes; (void)n_in; (void)out_size; (void)ws_size;

    const float* memory  = (const float*)d_in[0];
    const float* context = (const float*)d_in[1];
    const unsigned char* mask = (const unsigned char*)d_in[2];  // numpy bool = 1 byte
    const float* qp      = (const float*)d_in[3];
    const float* ctx_w   = (const float*)d_in[4];
    const float* ctx_b   = (const float*)d_in[5];
    const float* in_w    = (const float*)d_in[6];
    const float* in_b    = (const float*)d_in[7];
    const float* out_w   = (const float*)d_in[8];
    const float* out_b   = (const float*)d_in[9];
    const float* ln_g    = (const float*)d_in[10];
    const float* ln_b    = (const float*)d_in[11];
    const float* gate_w  = (const float*)d_in[12];
    const float* gate_b  = (const float*)d_in[13];

    float* out = (float*)d_out;
    float* out_read    = out;                                        // B*Q*D
    float* out_att     = out_read + (size_t)BB * QQ * DD;            // B*Q*M
    float* out_gates   = out_att + (size_t)BB * QQ * MM;             // B*Q
    float* out_queries = out_gates + (size_t)BB * QQ;                // B*Q*D

    char* ws = (char*)d_ws;
    bf16*  Kbuf = (bf16*)ws;
    bf16*  Vt   = (bf16*)(ws + ((size_t)64 << 20));
    float* SC   = (float*)(ws + ((size_t)128 << 20));
    bf16*  qs   = (bf16*)(ws + ((size_t)192 << 20));
    bf16*  ctxb = (bf16*)(ws + ((size_t)192 << 20) + ((size_t)1 << 19));
    bf16*  P    = Kbuf;   // K region reused for probabilities

    prep_kernel<<<BB, 256, 0, stream>>>(context, qp, ctx_w, ctx_b, in_w, in_b,
                                        gate_w, gate_b, out_gates, out_queries, qs);
    kv_proj_kernel<<<dim3(BB * MM / 128, 8), 256, KV_SMEM, stream>>>(memory, in_w, in_b,
                                                                     Kbuf, Vt);
    scores_kernel<<<dim3(MM / 512, BB * HH), 256, 0, stream>>>(Kbuf, qs, mask, SC);
    softmax_kernel<<<BB * HH, 256, 0, stream>>>(SC, P);
    attnmean_kernel<<<4096, 256, 0, stream>>>(P, out_att);
    ctx_kernel<<<BB * HH, 256, 0, stream>>>(P, Vt, ctxb);
    readout_kernel<<<(BB * QQ) / 16, 256, 0, stream>>>(ctxb, out_w, out_b, out_gates,
                                                       ln_g, ln_b, out_read);
}
